// MultiHopMSA_77850577207520
// MI455X (gfx1250) — matrix operand order repaired and checked by reading, not compiled, not run on hardware
//
#include <hip/hip_runtime.h>
#include <hip/hip_bf16.h>

// ---------------------------------------------------------------------------
// MultiHopMSA for MI455X (gfx1250, wave32).
// All matmuls via v_wmma_f32_16x16x32_bf16 (bf16 inputs, fp32 accumulate).
// - fragments are 2x b128 loads (layouts chosen so A and B operands are
//   contiguous per lane), outputs are b128 stores
// - f32<->bf16 via native v_cvt_pk_bf16_f32 ((__bf16) casts)
// - core GEMM: 32x64 wave tile (8 WMMA per K-step, B-fragments reused),
//   global_prefetch two K-steps ahead
//   B=2, N=2048, D=512, H=8, dk=64, HOPS=3
// ---------------------------------------------------------------------------

#define BB   2
#define NN   2048
#define DD   512
#define HH   8
#define DK   64
#define BH   (BB*HH)

typedef __attribute__((ext_vector_type(16))) __bf16 v16bf;
typedef __attribute__((ext_vector_type(8)))  __bf16 v8bf;
typedef __attribute__((ext_vector_type(8)))  float  v8f;

// ---- bf16 helpers: native converts ----------------------------------------
__device__ __forceinline__ unsigned short bfbits(float f) {
  union { __bf16 b; unsigned short s; } t; t.b = (__bf16)f; return t.s;
}
__device__ __forceinline__ float bits2f(unsigned short u) {
  union { unsigned short s; __bf16 b; } t; t.s = u; return (float)t.b;
}
union Pack8U { v8bf v; uint4 q; };
__device__ __forceinline__ uint4 pack8(const v8f f) {
  Pack8U u;
#pragma unroll
  for (int i = 0; i < 8; ++i) u.v[i] = (__bf16)f[i];   // -> v_cvt_pk_bf16_f32
  return u.q;
}
__device__ __forceinline__ uint4 pack8s(const float* __restrict__ p, float s) {
  Pack8U u;
#pragma unroll
  for (int i = 0; i < 8; ++i) u.v[i] = (__bf16)(p[i] * s);
  return u.q;
}

// ---- WMMA ------------------------------------------------------------------
__device__ __forceinline__ v8f wmma_bf16(v16bf a, v16bf b, v8f c) {
  return __builtin_amdgcn_wmma_f32_16x16x32_bf16(false, a, false, b,
                                                 (short)0, c, false, false);
}

// ---- fragment loaders ------------------------------------------------------
union FragU { v16bf v; uint4 q[2]; };

__device__ __forceinline__ v16bf load_frag2(const unsigned short* p0,
                                            const unsigned short* p1) {
  FragU u;
  u.q[0] = *(const uint4*)p0;
  u.q[1] = *(const uint4*)p1;
  return u.v;
}
// A-matrix 16x32: lanes 0-15 M=lane, K [0..15];
// lanes 16-31 same M, K [16..31]. Row-major bf16 source.
__device__ __forceinline__ v16bf load_a_bf16(const unsigned short* __restrict__ base,
                                             int ld, int k0) {
  const int lane = threadIdx.x & 31;
  const unsigned short* p = base + (long)(lane & 15) * ld + k0 + ((lane >> 4) << 4);
  return load_frag2(p, p + 8);
}
// B-matrix 32x16 from BT layout (BT[n][k] = B[k][n]).
__device__ __forceinline__ v16bf load_b_bt(const unsigned short* __restrict__ bt,
                                           int ldbt, int k0, int n0) {
  const int lane = threadIdx.x & 31;
  const unsigned short* p = bt + (long)(n0 + (lane & 15)) * ldbt + k0 + ((lane >> 4) << 4);
  return load_frag2(p, p + 8);
}
// f32 sources (kernel inputs), converted with native cvt (pairs pack free).
__device__ __forceinline__ v16bf load_a_f32(const float* __restrict__ base,
                                            int ld, int k0) {
  const int lane = threadIdx.x & 31;
  const float* p = base + (long)(lane & 15) * ld + k0 + ((lane >> 4) << 4);
  v16bf a;
#pragma unroll
  for (int i = 0; i < 8; ++i) a[i]     = (__bf16)p[i];
#pragma unroll
  for (int i = 0; i < 8; ++i) a[8 + i] = (__bf16)p[8 + i];
  return a;
}
__device__ __forceinline__ v16bf load_b_f32(const float* __restrict__ base,
                                            int ld, int k0, int n0) {
  const int lane = threadIdx.x & 31;
  const float* p = base + (long)(k0 + ((lane >> 4) << 4)) * ld + n0 + (lane & 15);
  v16bf b;
#pragma unroll
  for (int i = 0; i < 16; ++i) b[i] = (__bf16)p[(long)i * ld];
  return b;
}

// ---------------------------------------------------------------------------
// Kernel 1: QKV projection. X[4096,512]f32 @ W[512,1536]f32 ->
//   q,k row-major [B,H,N,dk], vT [B,H,dk,N] (packed b128 stores).
// ---------------------------------------------------------------------------
__global__ __launch_bounds__(128)
void mhmsa_qkv_kernel(const float* __restrict__ X, const float* __restrict__ W,
                      unsigned short* __restrict__ q,
                      unsigned short* __restrict__ k,
                      unsigned short* __restrict__ vT) {
  const int wave = threadIdx.x >> 5;
  const int m0 = blockIdx.x * 64 + wave * 16;
  const int n0 = blockIdx.y * 64;
  const float* Ab = X + (long)m0 * DD;
  v8f acc[4] = {};
  for (int k0 = 0; k0 < DD; k0 += 32) {
    v16bf a = load_a_f32(Ab, DD, k0);
#pragma unroll
    for (int j = 0; j < 4; ++j)
      acc[j] = wmma_bf16(a, load_b_f32(W, 3 * DD, k0, n0 + j * 16), acc[j]);
  }
  const int lane = threadIdx.x & 31;
  const int hi = lane >> 4, c0 = lane & 15;
  const int b = m0 >> 11, nn0 = (m0 & (NN - 1)) + hi * 8;
#pragma unroll
  for (int j = 0; j < 4; ++j) {
    const int col = n0 + j * 16 + c0;          // fixed per lane
    const int three = col >> 9;
    const int h = (col >> 6) & (HH - 1);
    const int d = col & (DK - 1);
    if (three == 2) {                          // v: transposed, packed b128
      *(uint4*)&vT[(((long)b * HH + h) * DK + d) * NN + nn0] = pack8(acc[j]);
    } else {                                   // q/k: row-major
      unsigned short* dst = (three == 0) ? q : k;
#pragma unroll
      for (int r = 0; r < 8; ++r)
        dst[(((long)b * HH + h) * NN + (nn0 + r)) * DK + d] = bfbits(acc[j][r]);
    }
  }
}

// ---------------------------------------------------------------------------
// Softmax helpers (one wave per 16-row strip; LDS = 16*2048 f32 + 16 inv).
// C-fragment element r <-> row r+8*(lane>=16), col = lane&15.
// ---------------------------------------------------------------------------
__device__ __forceinline__ void red_max16(float* mr) {
#pragma unroll
  for (int r = 0; r < 8; ++r)
#pragma unroll
    for (int msk = 1; msk < 16; msk <<= 1)
      mr[r] = fmaxf(mr[r], __shfl_xor(mr[r], msk));
}
__device__ __forceinline__ void expsum_write_inv(float* __restrict__ smem,
                                                 float* __restrict__ srow,
                                                 const float* __restrict__ mr) {
  const int lane = threadIdx.x & 31;
  const int hi = lane >> 4, c0 = lane & 15;
  float sr[8];
#pragma unroll
  for (int r = 0; r < 8; ++r) sr[r] = 0.f;
  for (int ct = 0; ct < NN / 16; ++ct) {
#pragma unroll
    for (int r = 0; r < 8; ++r) {
      float* p = &smem[(r + hi * 8) * NN + ct * 16 + c0];
      float e = __expf(*p - mr[r]);
      *p = e; sr[r] += e;
    }
  }
#pragma unroll
  for (int r = 0; r < 8; ++r)
#pragma unroll
    for (int msk = 1; msk < 16; msk <<= 1)
      sr[r] += __shfl_xor(sr[r], msk);
  if (c0 == 0) {
#pragma unroll
    for (int r = 0; r < 8; ++r) srow[hi * 8 + r] = 1.0f / sr[r];
  }
  __syncthreads();
}
// Chunked normalize+store: each lane handles 8 contiguous cols -> b128 stores.
__device__ __forceinline__ void store_rowmajor(const float* __restrict__ smem,
                                               const float* __restrict__ srow,
                                               unsigned short* __restrict__ A,
                                               long aoff, int m0) {
  const int lane = threadIdx.x & 31;
  for (int i = 0; i < (16 * NN / 8) / 32; ++i) {
    const int t = i * 32 + lane;
    const int rr = t >> 8, cb = t & 255;
    *(uint4*)&A[aoff + (long)(m0 + rr) * NN + cb * 8] =
        pack8s(&smem[rr * NN + cb * 8], srow[rr]);
  }
}
// Transposed copy (for A2 as B-operand): AT[col][m0..m0+15], 2x b128 per col.
__device__ __forceinline__ void store_transposed(const float* __restrict__ smem,
                                                 const float* __restrict__ srow,
                                                 unsigned short* __restrict__ AT,
                                                 long aoff, int m0) {
  const int lane = threadIdx.x & 31;
  for (int col = lane; col < NN; col += 32) {
    Pack8U lo, hiu;
#pragma unroll
    for (int rr = 0; rr < 8; ++rr)
      lo.v[rr]  = (__bf16)(smem[rr * NN + col] * srow[rr]);
#pragma unroll
    for (int rr = 0; rr < 8; ++rr)
      hiu.v[rr] = (__bf16)(smem[(rr + 8) * NN + col] * srow[rr + 8]);
    uint4* dst = (uint4*)&AT[aoff + (long)col * NN + m0];
    dst[0] = lo.q;
    dst[1] = hiu.q;
  }
}

// ---------------------------------------------------------------------------
// Kernel 2: scores + softmax.
// mode 0: A1 = softmax(s*q1k1ᵀ) (row-major), A2 likewise + transposed copy.
// mode 1: A = softmax(0.5*S1 + 0.75*S2 + 0.5*logaddexp(S1,S2) + 0.5*log(C2+eps))
// ---------------------------------------------------------------------------
__global__ __launch_bounds__(32)
void mhmsa_attn_kernel(const unsigned short* __restrict__ q1,
                       const unsigned short* __restrict__ k1,
                       const unsigned short* __restrict__ q2,
                       const unsigned short* __restrict__ k2,
                       const unsigned short* __restrict__ C2,
                       unsigned short* __restrict__ out1,
                       unsigned short* __restrict__ out2,
                       unsigned short* __restrict__ out2T,
                       int mode) {
  extern __shared__ float smem[];                 // 16*NN f32 + 16 inv
  float* srow = smem + 16 * NN;
  const int bh = blockIdx.y;
  const int m0 = blockIdx.x * 16;
  const int lane = threadIdx.x & 31;
  const int hi = lane >> 4, c0 = lane & 15;
  const long qoff = (long)bh * NN * DK;
  const long aoff = (long)bh * NN * NN;
  const float scale = 0.125f;                     // 1/sqrt(dk)

  if (mode == 0) {
    for (int mat = 0; mat < 2; ++mat) {
      const unsigned short* q = mat ? q2 : q1;
      const unsigned short* k = mat ? k2 : k1;    // row-major k == Bᵀ of qkᵀ
      const v16bf aq0 = load_a_bf16(q + qoff + (long)m0 * DK, DK, 0);
      const v16bf aq1 = load_a_bf16(q + qoff + (long)m0 * DK, DK, 32);
      float mr[8];
#pragma unroll
      for (int r = 0; r < 8; ++r) mr[r] = -3.402823466e38f;
      for (int ct = 0; ct < NN / 16; ++ct) {
        v8f acc = {};
        acc = wmma_bf16(aq0, load_b_bt(k + qoff, DK, 0,  ct * 16), acc);
        acc = wmma_bf16(aq1, load_b_bt(k + qoff, DK, 32, ct * 16), acc);
#pragma unroll
        for (int r = 0; r < 8; ++r) {
          const float s = acc[r] * scale;
          smem[(r + hi * 8) * NN + ct * 16 + c0] = s;
          mr[r] = fmaxf(mr[r], s);
        }
      }
      red_max16(mr);
      expsum_write_inv(smem, srow, mr);
      store_rowmajor(smem, srow, mat ? out2 : out1, aoff, m0);
      if (mat) store_transposed(smem, srow, out2T, aoff, m0);
      __syncthreads();
    }
  } else {
    const v16bf a10 = load_a_bf16(q1 + qoff + (long)m0 * DK, DK, 0);
    const v16bf a11 = load_a_bf16(q1 + qoff + (long)m0 * DK, DK, 32);
    const v16bf a20 = load_a_bf16(q2 + qoff + (long)m0 * DK, DK, 0);
    const v16bf a21 = load_a_bf16(q2 + qoff + (long)m0 * DK, DK, 32);
    for (int ct = 0; ct < NN / 16; ++ct) {
      v8f acc1 = {}, acc2 = {};
      acc1 = wmma_bf16(a10, load_b_bt(k1 + qoff, DK, 0,  ct * 16), acc1);
      acc1 = wmma_bf16(a11, load_b_bt(k1 + qoff, DK, 32, ct * 16), acc1);
      acc2 = wmma_bf16(a20, load_b_bt(k2 + qoff, DK, 0,  ct * 16), acc2);
      acc2 = wmma_bf16(a21, load_b_bt(k2 + qoff, DK, 32, ct * 16), acc2);
#pragma unroll
      for (int r = 0; r < 8; ++r) {
        const float s1 = acc1[r] * scale;
        const float s2 = acc2[r] * scale;
        const float lae = fmaxf(s1, s2) + __logf(1.0f + __expf(-fabsf(s1 - s2)));
        smem[(r + hi * 8) * NN + ct * 16 + c0] = 0.5f * s1 + 0.75f * s2 + 0.5f * lae;
      }
    }
    __syncthreads();
    // chunked: add 0.5*log(C2+eps) with packed b128 C2 loads
    for (int i = 0; i < (16 * NN / 8) / 32; ++i) {
      const int t = i * 32 + lane;
      const int rr = t >> 8, cb = t & 255;
      const uint4 cc = *(const uint4*)&C2[aoff + (long)(m0 + rr) * NN + cb * 8];
      const unsigned short* cs = (const unsigned short*)&cc;
      float* p = &smem[rr * NN + cb * 8];
#pragma unroll
      for (int e = 0; e < 8; ++e)
        p[e] += 0.5f * __logf(bits2f(cs[e]) + 1e-6f);
    }
    __syncthreads();
    float mr[8];
#pragma unroll
    for (int r = 0; r < 8; ++r) mr[r] = -3.402823466e38f;
    for (int ct = 0; ct < NN / 16; ++ct)
#pragma unroll
      for (int r = 0; r < 8; ++r)
        mr[r] = fmaxf(mr[r], smem[(r + hi * 8) * NN + ct * 16 + c0]);
    red_max16(mr);
    expsum_write_inv(smem, srow, mr);
    store_rowmajor(smem, srow, out1, aoff, m0);
  }
}

// ---------------------------------------------------------------------------
// Kernel 3: batched bf16 GEMM  C[b] = A[b] @ B[b], B given in BT layout.
// Block 128 = 4 waves; wave tile 32x64 (B-fragments reused for 2 row strips
// -> 8 WMMA per K-step); grid (M/128, N/64, batch).
// trans==0: row-major out via 16KB LDS staging (coalesced b128 stores).
// trans==1: transposed-packed out (CT[col][row]), direct b128 stores.
// global_prefetch (via __builtin_prefetch) of A/B lines two K-steps ahead.
// ---------------------------------------------------------------------------
__global__ __launch_bounds__(128)
void mhmsa_gemm_bf16_kernel(const unsigned short* __restrict__ A, long sA,
                            const unsigned short* __restrict__ BT, long sB,
                            unsigned short* __restrict__ C, long sC,
                            int K, int lda, int ldbt, int ldc, int trans) {
  __shared__ alignas(16) unsigned short stg[4 * 32 * 64];   // 16 KB
  const int wave = threadIdx.x >> 5;
  const int lane = threadIdx.x & 31;
  const int m0 = blockIdx.x * 128 + wave * 32;
  const int n0 = blockIdx.y * 64;
  const int bz = blockIdx.z;
  const unsigned short* Ab = A + (long)bz * sA + (long)m0 * lda;
  const unsigned short* Bb = BT + (long)bz * sB;
  v8f acc[2][4] = {};
#pragma unroll 2
  for (int k0 = 0; k0 < K; k0 += 32) {
    v16bf a0 = load_a_bf16(Ab, lda, k0);
    v16bf a1 = load_a_bf16(Ab + (long)16 * lda, lda, k0);
    if (k0 + 64 < K) {                            // prefetch two K-steps ahead
      __builtin_prefetch(Ab + (long)lane * lda + k0 + 64, 0, 0);
      __builtin_prefetch(Bb + (long)(n0 + (lane & 31)) * ldbt + k0 + 64, 0, 0);
    }
#pragma unroll
    for (int j = 0; j < 4; ++j) {
      v16bf b = load_b_bt(Bb, ldbt, k0, n0 + j * 16);
      acc[0][j] = wmma_bf16(a0, b, acc[0][j]);
      acc[1][j] = wmma_bf16(a1, b, acc[1][j]);
    }
  }
  const int hi = lane >> 4, c0 = lane & 15;
  unsigned short* Cbp = C + (long)bz * sC;
  if (trans) {                                    // CT[col][row]: packed direct
#pragma unroll
    for (int st = 0; st < 2; ++st)
#pragma unroll
      for (int j = 0; j < 4; ++j)
        *(uint4*)&Cbp[(long)(n0 + j * 16 + c0) * ldc + m0 + st * 16 + hi * 8] =
            pack8(acc[st][j]);
  } else {                                        // stage in LDS, b128 out
    unsigned short* sg = &stg[wave * 32 * 64];
#pragma unroll
    for (int st = 0; st < 2; ++st)
#pragma unroll
      for (int j = 0; j < 4; ++j)
#pragma unroll
        for (int r = 0; r < 8; ++r)
          sg[(st * 16 + r + hi * 8) * 64 + j * 16 + c0] = bfbits(acc[st][j][r]);
#pragma unroll
    for (int i = 0; i < 8; ++i) {                 // 32*64/8 = 256 chunks / 32
      const int t = i * 32 + lane;
      const int rr = t >> 3, cb = t & 7;
      *(uint4*)&Cbp[(long)(m0 + rr) * ldc + n0 + cb * 8] =
          *(const uint4*)&sg[rr * 64 + cb * 8];
    }
  }
}

// ---------------------------------------------------------------------------
// Kernel 4: y = yA + sigmoid(chain_logit)*y_chain, [B,H,N,dk] -> [B,N,D] bf16
// Vectorized over 8-element chunks (b128 loads/stores).
// ---------------------------------------------------------------------------
__global__ __launch_bounds__(256)
void mhmsa_combine_kernel(const unsigned short* __restrict__ yA,
                          const unsigned short* __restrict__ yC,
                          const float* __restrict__ chain_logit,
                          unsigned short* __restrict__ y) {
  const int i = blockIdx.x * blockDim.x + threadIdx.x;   // chunk of 8
  if (i >= BB * HH * NN * DK / 8) return;
  const long e0 = (long)i * 8;
  const int d0 = (int)(e0 & (DK - 1));
  const int n  = (int)((e0 >> 6) & (NN - 1));
  const int h  = (int)((e0 >> 17) & (HH - 1));
  const int b  = (int)(e0 >> 20);
  const float w = 1.0f / (1.0f + __expf(-chain_logit[0]));
  const uint4 av = *(const uint4*)&yA[e0];
  const uint4 cv = *(const uint4*)&yC[e0];
  const unsigned short* as = (const unsigned short*)&av;
  const unsigned short* cs = (const unsigned short*)&cv;
  Pack8U o;
#pragma unroll
  for (int e = 0; e < 8; ++e)
    o.v[e] = (__bf16)(bits2f(as[e]) + w * bits2f(cs[e]));
  *(uint4*)&y[((long)(b * NN + n)) * DD + h * DK + d0] = o.q;
}

// ---------------------------------------------------------------------------
// Kernel 5: projection. y[4096,512]bf16 @ Wproj[512,512]f32 -> out f32
// ---------------------------------------------------------------------------
__global__ __launch_bounds__(128)
void mhmsa_proj_kernel(const unsigned short* __restrict__ Y,
                       const float* __restrict__ W,
                       float* __restrict__ out) {
  const int wave = threadIdx.x >> 5;
  const int m0 = blockIdx.x * 64 + wave * 16;
  const int n0 = blockIdx.y * 64;
  const unsigned short* Ab = Y + (long)m0 * DD;
  v8f acc[4] = {};
  for (int k0 = 0; k0 < DD; k0 += 32) {
    v16bf a = load_a_bf16(Ab, DD, k0);
#pragma unroll
    for (int j = 0; j < 4; ++j)
      acc[j] = wmma_bf16(a, load_b_f32(W, DD, k0, n0 + j * 16), acc[j]);
  }
  const int lane = threadIdx.x & 31;
  const int hi = lane >> 4, c0 = lane & 15;
#pragma unroll
  for (int j = 0; j < 4; ++j)
#pragma unroll
    for (int r = 0; r < 8; ++r)
      out[(long)(m0 + r + hi * 8) * DD + n0 + j * 16 + c0] = acc[j][r];
}

// ---------------------------------------------------------------------------
// Host: workspace layout + launch sequence (all on `stream`).
// ---------------------------------------------------------------------------
extern "C" void kernel_launch(void* const* d_in, const int* in_sizes, int n_in,
                              void* d_out, int out_size, void* d_ws, size_t ws_size,
                              hipStream_t stream) {
  (void)in_sizes; (void)n_in; (void)out_size;
  const float* x      = (const float*)d_in[0];
  const float* Wqkv1  = (const float*)d_in[1];
  const float* Wqkv2  = (const float*)d_in[2];
  const float* Wproj  = (const float*)d_in[3];
  const float* clogit = (const float*)d_in[4];

  const size_t QKV_BYTES = (size_t)BB * HH * NN * DK * 2;   // 4 MiB
  const size_t ATT_BYTES = (size_t)BB * HH * NN * NN * 2;   // 128 MiB
  const long   QKV_S = (long)NN * DK;
  const long   ATT_S = (long)NN * NN;

  char* w = (char*)d_ws;
  size_t off = 0;
  auto take = [&](size_t bytes) { char* p = w + off; off += (bytes + 255) & ~(size_t)255; return p; };
  unsigned short* q1   = (unsigned short*)take(QKV_BYTES);
  unsigned short* k1   = (unsigned short*)take(QKV_BYTES);   // row-major == Bᵀ
  unsigned short* v1T  = (unsigned short*)take(QKV_BYTES);   // [B,H,dk,N]
  unsigned short* q2   = (unsigned short*)take(QKV_BYTES);
  unsigned short* k2   = (unsigned short*)take(QKV_BYTES);
  unsigned short* v2T  = (unsigned short*)take(QKV_BYTES);
  unsigned short* A1   = (unsigned short*)take(ATT_BYTES);   // A-operand
  unsigned short* A2   = (unsigned short*)take(ATT_BYTES);   // A-operand (hops)
  unsigned short* A2T  = (unsigned short*)take(ATT_BYTES);   // B-operand (chain)
  unsigned short* Cb   = (unsigned short*)take(ATT_BYTES);   // C, then A
  unsigned short* C2   = (unsigned short*)take(ATT_BYTES);
  unsigned short* t1T  = (unsigned short*)take(QKV_BYTES);   // [dk,N] per bh
  unsigned short* t2T  = (unsigned short*)take(QKV_BYTES);
  unsigned short* ych  = (unsigned short*)take(QKV_BYTES);   // row-major
  unsigned short* yA   = (unsigned short*)take(QKV_BYTES);
  unsigned short* ybuf = (unsigned short*)take(QKV_BYTES);
  if (off > ws_size) return;

  // 1) QKV projections
  mhmsa_qkv_kernel<<<dim3((BB * NN) / 64, (3 * DD) / 64), 128, 0, stream>>>(x, Wqkv1, q1, k1, v1T);
  mhmsa_qkv_kernel<<<dim3((BB * NN) / 64, (3 * DD) / 64), 128, 0, stream>>>(x, Wqkv2, q2, k2, v2T);

  // 2) A1, A2 (+A2ᵀ)
  const size_t lds = (size_t)16 * NN * sizeof(float) + 64;
  mhmsa_attn_kernel<<<dim3(NN / 16, BH), 32, lds, stream>>>(
      q1, k1, q2, k2, nullptr, A1, A2, A2T, 0);

  // 3) C = A1 @ A2 ; C2 = C @ A2   (549-GFLOP core; B-side from A2ᵀ)
  mhmsa_gemm_bf16_kernel<<<dim3(NN / 128, NN / 64, BH), 128, 0, stream>>>(
      A1, ATT_S, A2T, ATT_S, Cb, ATT_S, NN, NN, NN, NN, 0);
  mhmsa_gemm_bf16_kernel<<<dim3(NN / 128, NN / 64, BH), 128, 0, stream>>>(
      Cb, ATT_S, A2T, ATT_S, C2, ATT_S, NN, NN, NN, NN, 0);

  // 4) A = softmax(Smix + 0.5*log(C2+eps)) -> Cb (A-operand layout)
  mhmsa_attn_kernel<<<dim3(NN / 16, BH), 32, lds, stream>>>(
      q1, k1, q2, k2, C2, Cb, nullptr, nullptr, 1);

  // 5) transport hops (transposed-packed outputs keep B-side contiguous)
  mhmsa_gemm_bf16_kernel<<<dim3(NN / 128, 1, BH), 128, 0, stream>>>(
      A2, ATT_S, v2T, QKV_S, t1T, QKV_S, NN, NN, NN, NN, 1);
  mhmsa_gemm_bf16_kernel<<<dim3(NN / 128, 1, BH), 128, 0, stream>>>(
      A2, ATT_S, t1T, QKV_S, t2T, QKV_S, NN, NN, NN, NN, 1);
  mhmsa_gemm_bf16_kernel<<<dim3(NN / 128, 1, BH), 128, 0, stream>>>(
      A1, ATT_S, t2T, QKV_S, ych, QKV_S, NN, NN, NN, DK, 0);
  mhmsa_gemm_bf16_kernel<<<dim3(NN / 128, 1, BH), 128, 0, stream>>>(
      Cb, ATT_S, v1T, QKV_S, yA, QKV_S, NN, NN, NN, DK, 0);

  // 6) combine + projection
  mhmsa_combine_kernel<<<(BB * HH * NN * DK / 8 + 255) / 256, 256, 0, stream>>>(
      yA, ych, clogit, ybuf);
  mhmsa_proj_kernel<<<dim3((BB * NN) / 64, DD / 64), 128, 0, stream>>>(
      ybuf, Wproj, (float*)d_out);
}
